// Attention_25074019074176
// MI455X (gfx1250) — compile-verified, matrix operand order checked by reading, not run on hardware
//
#include <hip/hip_runtime.h>

// ---------------------------------------------------------------- types
typedef __attribute__((ext_vector_type(16))) _Float16 v16h;
typedef __attribute__((ext_vector_type(8)))  float    v8f;
typedef __attribute__((ext_vector_type(4)))  unsigned int u32x4;
typedef __attribute__((ext_vector_type(4)))  float    f32x4;
typedef int v4i __attribute__((vector_size(16)));   // matches builtin param type

union HFrag { v16h v; u32x4 u[2]; _Float16 h[16]; };

#define BATCH 4
#define SEQ   4096
#define NH    16
#define HD    128
#define HID   2048
#define N3    6144

// WMMA 16x16x32 f16 -> f32 accumulate
#define WMMA_F16(a, b, c) \
  __builtin_amdgcn_wmma_f32_16x16x32_f16(false, (a), false, (b), (short)0, (c), false, false)

// ---- CDNA5 async global->LDS copy (16 bytes), tracked by ASYNCcnt ----------
__device__ __forceinline__ void async_cp16(const _Float16* g, _Float16* l) {
#if __has_builtin(__builtin_amdgcn_global_load_async_to_lds_b128)
  __builtin_amdgcn_global_load_async_to_lds_b128(
      (__attribute__((address_space(1))) v4i*)(v4i*)(void*)g,
      (__attribute__((address_space(3))) v4i*)(v4i*)(void*)l,
      0, 0);
#else
  const unsigned loff =
      (unsigned)(size_t)(__attribute__((address_space(3))) const void*)l;
  asm volatile("global_load_async_to_lds_b128 %0, %1, off"
               :: "v"(loff), "v"(g) : "memory");
#endif
}

#if __has_builtin(__builtin_amdgcn_s_wait_asynccnt)
#define WAIT_ASYNC(n) __builtin_amdgcn_s_wait_asynccnt(n)
#else
#define WAIT_ASYNC(n) asm volatile("s_wait_asynccnt %0" :: "i"(n) : "memory")
#endif

// Load a 16x32 (f16) A/B fragment from LDS (ISA 7.12.2 layout):
// lanes 0-15 row m=l, halves K[0..7] then K[16..23]; lanes 16-31 K[8..15]/K[24..31].
// B fragments use the same pattern on an [N][K]-major tile.
__device__ __forceinline__ v16h ld_frag(const _Float16* base, int ld, int lane) {
  const int l = lane & 15, hi = lane >> 4;
  const _Float16* p = base + l * ld + hi * 8;
  HFrag f;
  f.u[0] = *(const u32x4*)(p);
  f.u[1] = *(const u32x4*)(p + 16);
  return f.v;
}

// ----------------------------------------------------- x fp32 -> f16 convert
__global__ __launch_bounds__(256) void convert_x(
    const float* __restrict__ in, _Float16* __restrict__ out) {
  const size_t i = ((size_t)blockIdx.x * 256 + threadIdx.x) * 8;
  f32x4 a0 = *(const f32x4*)(in + i);
  f32x4 a1 = *(const f32x4*)(in + i + 4);
  union { _Float16 h[8]; u32x4 u; } t;
  for (int j = 0; j < 4; ++j) { t.h[j] = (_Float16)a0[j]; t.h[4 + j] = (_Float16)a1[j]; }
  *(u32x4*)(out + i) = t.u;
}

// ------------------------------------------------- weight convert+transpose
// in: [rows][cols] fp32  ->  out: [cols][rows] f16
__global__ __launch_bounds__(256) void convert_transpose(
    const float* __restrict__ in, _Float16* __restrict__ out, int rows, int cols) {
  __shared__ float tile[32][33];
  const int bx = blockIdx.x * 32;
  const int by = blockIdx.y * 32;
  const int tx = threadIdx.x, ty = threadIdx.y;   // (32, 8)
  for (int i = 0; i < 32; i += 8)
    tile[ty + i][tx] = in[(size_t)(by + ty + i) * cols + bx + tx];
  __syncthreads();
  for (int i = 0; i < 32; i += 8)
    out[(size_t)(bx + ty + i) * rows + by + tx] = (_Float16)tile[tx][ty + i];
}

// ---------------------------------------------------------------- QKV GEMM
// [16384 x 2048] f16 @ WqkvT -> +bias -> scatter f16 into Q/K/V [B, H, S, D].
// Double-buffered async global->LDS pipeline.
__global__ __launch_bounds__(256) void qkv_gemm(
    const _Float16* __restrict__ xh, const _Float16* __restrict__ WT,
    const float* __restrict__ bias,
    _Float16* __restrict__ qh, _Float16* __restrict__ kh, _Float16* __restrict__ vh) {
  constexpr int LDT = 40;                       // 32 + 8 pad: conflict-free b128
  __shared__ _Float16 As[2][128 * LDT];
  __shared__ _Float16 Bs[2][128 * LDT];
  const int tid = threadIdx.x, lane = tid & 31, wave = tid >> 5;
  const int row0 = blockIdx.y * 128;
  const int col0 = blockIdx.x * 128;
  const int wm = wave & 3, wn = wave >> 2;      // wave tile: 32(M) x 64(N)

  v8f acc[2][4];
  for (int i = 0; i < 2; ++i) for (int j = 0; j < 4; ++j) acc[i][j] = {};

  const int lrow = tid >> 1, lseg = tid & 1;    // 128 rows x 2 x 16-half segs
  const _Float16* asrc = xh + (size_t)(row0 + lrow) * HID + lseg * 16;
  const _Float16* bsrc = WT + (size_t)(col0 + lrow) * HID + lseg * 16;

  auto issue = [&](int k0, int buf) {
    _Float16* a = &As[buf][lrow * LDT + lseg * 16];
    _Float16* b = &Bs[buf][lrow * LDT + lseg * 16];
    async_cp16(asrc + k0,     a);
    async_cp16(asrc + k0 + 8, a + 8);
    async_cp16(bsrc + k0,     b);
    async_cp16(bsrc + k0 + 8, b + 8);
  };

  issue(0, 0);
  for (int k0 = 0, it = 0; k0 < HID; k0 += 32, ++it) {
    const int cur = it & 1;
    if (k0 + 32 < HID) {
      issue(k0 + 32, cur ^ 1);
      if (k0 + 64 < HID) {
        __builtin_prefetch(asrc + k0 + 64, 0, 3);   // warm L2 for tile k+2
        __builtin_prefetch(bsrc + k0 + 64, 0, 3);
      }
      WAIT_ASYNC(4);          // in-order: tile k fully landed in LDS
    } else {
      WAIT_ASYNC(0);
    }
    __syncthreads();

    v16h af[2], bf[4];
    for (int i = 0; i < 2; ++i) af[i] = ld_frag(&As[cur][(wm * 32 + i * 16) * LDT], LDT, lane);
    for (int j = 0; j < 4; ++j) bf[j] = ld_frag(&Bs[cur][(wn * 64 + j * 16) * LDT], LDT, lane);
    for (int i = 0; i < 2; ++i)
      for (int j = 0; j < 4; ++j)
        acc[i][j] = WMMA_F16(af[i], bf[j], acc[i][j]);
    __syncthreads();          // readers done before buf is async-overwritten
  }

  const int l = lane & 15, hi = lane >> 4;
  for (int i = 0; i < 2; ++i)
    for (int j = 0; j < 4; ++j) {
      const int n = col0 + wn * 64 + j * 16 + l;
      const float bv = bias[n];
      const int which = n >> 11, h = (n & 2047) >> 7, d = n & 127;
      _Float16* dst = (which == 0) ? qh : (which == 1) ? kh : vh;
      for (int r = 0; r < 8; ++r) {
        const int m = row0 + wm * 32 + i * 16 + hi * 8 + r;
        const int bb = m >> 12, ss = m & 4095;
        dst[((((size_t)bb * NH) + h) * SEQ + ss) * HD + d] = (_Float16)(acc[i][j][r] + bv);
      }
    }
}

// ------------------------------------------------------------ flash attention
// grid (S/128, NH, B); 8 waves x 16 query rows each; 64-key chunks streamed.
__global__ __launch_bounds__(256) void attn(
    const _Float16* __restrict__ qh, const _Float16* __restrict__ kh,
    const _Float16* __restrict__ vh, _Float16* __restrict__ co) {
  constexpr int LDK = 136;   // 128 + 8 pad
  constexpr int LDV = 72;    //  64 + 8 pad
  constexpr int LDP = 72;
  __shared__ _Float16 Ks[64 * LDK];       // K chunk [key][d]   (async-filled)
  __shared__ _Float16 Vt[128 * LDV];      // V chunk [d][key]   (transposed)
  __shared__ _Float16 Ps[8][16 * LDP];    // per-wave P staging [row][key]

  const int tid = threadIdx.x, lane = tid & 31, wave = tid >> 5;
  const int b = blockIdx.z, h = blockIdx.y;
  const int q0 = blockIdx.x * 128 + wave * 16;
  const size_t head = ((size_t)b * NH + h) * SEQ;
  const _Float16* qb = qh + head * HD;
  const _Float16* kb = kh + head * HD;
  const _Float16* vb = vh + head * HD;

  // Q fragments (A layout) in registers for the whole pass: 4 k-steps of 32
  v16h qf[4];
  {
    const int l = lane & 15, hi = lane >> 4;
    const _Float16* qp = qb + (size_t)(q0 + l) * HD + hi * 8;
    for (int kk = 0; kk < 4; ++kk) {
      HFrag f;
      f.u[0] = *(const u32x4*)(qp + kk * 32);
      f.u[1] = *(const u32x4*)(qp + kk * 32 + 16);
      qf[kk] = f.v;
    }
  }

  v8f cacc[8];
  for (int nt = 0; nt < 8; ++nt) cacc[nt] = {};
  float mrow[8], lsum[8];
  for (int r = 0; r < 8; ++r) { mrow[r] = -1e30f; lsum[r] = 0.f; }
  const float scale = 0.08838834764831845f;   // 1/sqrt(128)

  const int kr = tid >> 2, seg = tid & 3;     // 64 rows x 4 x 32-half segments
  const _Float16* ksrc = kb + (size_t)kr * HD + seg * 32;
  const _Float16* vsrc0 = vb + (size_t)kr * HD + seg * 32;
  _Float16* kdst = &Ks[kr * LDK + seg * 32];

  for (int j0 = 0; j0 < SEQ; j0 += 64) {
    // K chunk: async global->LDS (4 x b128 per thread)
    for (int c = 0; c < 4; ++c)
      async_cp16(ksrc + (size_t)j0 * HD + c * 8, kdst + c * 8);

    // V chunk: load + transpose-scatter into Vt
    {
      const _Float16* vsrc = vsrc0 + (size_t)j0 * HD;
      HFrag fv0, fv1;
      fv0.u[0] = *(const u32x4*)(vsrc);      fv0.u[1] = *(const u32x4*)(vsrc + 8);
      fv1.u[0] = *(const u32x4*)(vsrc + 16); fv1.u[1] = *(const u32x4*)(vsrc + 24);
      for (int d = 0; d < 16; ++d) Vt[(seg * 32 + d) * LDV + kr]      = fv0.h[d];
      for (int d = 0; d < 16; ++d) Vt[(seg * 32 + 16 + d) * LDV + kr] = fv1.h[d];
    }
    WAIT_ASYNC(0);
    __syncthreads();

    // scores: 16x64 = Q(16x128) @ K^T -> 4 n-tiles x 4 k-steps of WMMA
    v8f sacc[4];
    for (int t = 0; t < 4; ++t) sacc[t] = {};
    for (int t = 0; t < 4; ++t)
      for (int kk = 0; kk < 4; ++kk) {
        v16h bf = ld_frag(&Ks[(t * 16) * LDK + kk * 32], LDK, lane);
        sacc[t] = WMMA_F16(qf[kk], bf, sacc[t]);
      }

    // streaming softmax (rows across 16-lane halves; shfl_xor reductions)
    for (int r = 0; r < 8; ++r) {
      float sv[4], mx = -1e30f;
      for (int t = 0; t < 4; ++t) { sv[t] = sacc[t][r] * scale; mx = fmaxf(mx, sv[t]); }
      for (int off = 8; off >= 1; off >>= 1) mx = fmaxf(mx, __shfl_xor(mx, off, 32));
      const float mnew = fmaxf(mrow[r], mx);
      const float corr = __expf(mrow[r] - mnew);
      float sum = 0.f;
      for (int t = 0; t < 4; ++t) {
        const float p = __expf(sv[t] - mnew);
        sacc[t][r] = p;
        sum += p;
      }
      for (int off = 8; off >= 1; off >>= 1) sum += __shfl_xor(sum, off, 32);
      lsum[r] = lsum[r] * corr + sum;
      mrow[r] = mnew;
      for (int nt = 0; nt < 8; ++nt) cacc[nt][r] = cacc[nt][r] * corr;
    }

    // stage P (f16) through wave-private LDS: C layout -> A layout.
    // LDS ops from the same wave complete in order (ISA 7.3): no barrier needed.
    {
      const int l = lane & 15, hi = lane >> 4;
      for (int t = 0; t < 4; ++t)
        for (int r = 0; r < 8; ++r)
          Ps[wave][(hi * 8 + r) * LDP + t * 16 + l] = (_Float16)sacc[t][r];
    }
    v16h pf[2];
    for (int kk = 0; kk < 2; ++kk) pf[kk] = ld_frag(&Ps[wave][kk * 32], LDP, lane);

    // context += P(16x64) @ V(64x128): 8 n-tiles x 2 k-steps
    for (int nt = 0; nt < 8; ++nt)
      for (int kk = 0; kk < 2; ++kk) {
        v16h bf = ld_frag(&Vt[(nt * 16) * LDV + kk * 32], LDV, lane);
        cacc[nt] = WMMA_F16(pf[kk], bf, cacc[nt]);
      }
    __syncthreads();   // readers done before Ks/Vt are refilled
  }

  // finalize: ctx /= l, store f16 into [B, H, S, D]
  {
    const int l = lane & 15, hi = lane >> 4;
    _Float16* ob = co + (head + q0) * HD;
    for (int r = 0; r < 8; ++r) {
      const float inv = 1.0f / lsum[r];
      for (int nt = 0; nt < 8; ++nt)
        ob[(size_t)(hi * 8 + r) * HD + nt * 16 + l] = (_Float16)(cacc[nt][r] * inv);
    }
  }
}

// ---------------------------------------------------------------- out proj
// out[16384 x 2048] fp32 = ctx(f16, head-transposed gather) @ WprojT + bias
__global__ __launch_bounds__(256) void proj_gemm(
    const _Float16* __restrict__ ctx, const _Float16* __restrict__ WT,
    const float* __restrict__ bias, float* __restrict__ out) {
  constexpr int LDT = 40;
  __shared__ _Float16 As[2][128 * LDT];
  __shared__ _Float16 Bs[2][128 * LDT];
  const int tid = threadIdx.x, lane = tid & 31, wave = tid >> 5;
  const int row0 = blockIdx.y * 128;
  const int col0 = blockIdx.x * 128;
  const int wm = wave & 3, wn = wave >> 2;

  v8f acc[2][4];
  for (int i = 0; i < 2; ++i) for (int j = 0; j < 4; ++j) acc[i][j] = {};

  const int lrow = tid >> 1, lseg = tid & 1;
  const int grow = row0 + lrow, bb = grow >> 12, ss = grow & 4095;
  const _Float16* wsrc = WT + (size_t)(col0 + lrow) * HID + lseg * 16;

  auto issue = [&](int k0, int buf) {
    // context col k = h*128 + d (transpose(0,2,1,3) fused into the gather)
    const int k = k0 + lseg * 16;
    const _Float16* asrc =
        ctx + ((((size_t)bb * NH) + (k >> 7)) * SEQ + ss) * HD + (k & 127);
    _Float16* a = &As[buf][lrow * LDT + lseg * 16];
    _Float16* b = &Bs[buf][lrow * LDT + lseg * 16];
    async_cp16(asrc,         a);
    async_cp16(asrc + 8,     a + 8);
    async_cp16(wsrc + k0,     b);
    async_cp16(wsrc + k0 + 8, b + 8);
  };

  issue(0, 0);
  for (int k0 = 0, it = 0; k0 < HID; k0 += 32, ++it) {
    const int cur = it & 1;
    if (k0 + 32 < HID) {
      issue(k0 + 32, cur ^ 1);
      if (k0 + 64 < HID) __builtin_prefetch(wsrc + k0 + 64, 0, 3);
      WAIT_ASYNC(4);
    } else {
      WAIT_ASYNC(0);
    }
    __syncthreads();

    v16h af[2], bf[4];
    for (int i = 0; i < 2; ++i) af[i] = ld_frag(&As[cur][(wm * 32 + i * 16) * LDT], LDT, lane);
    for (int j = 0; j < 4; ++j) bf[j] = ld_frag(&Bs[cur][(wn * 64 + j * 16) * LDT], LDT, lane);
    for (int i = 0; i < 2; ++i)
      for (int j = 0; j < 4; ++j)
        acc[i][j] = WMMA_F16(af[i], bf[j], acc[i][j]);
    __syncthreads();
  }

  const int l = lane & 15, hi = lane >> 4;
  for (int i = 0; i < 2; ++i)
    for (int j = 0; j < 4; ++j) {
      const int n = col0 + wn * 64 + j * 16 + l;
      const float bv = bias[n];
      for (int r = 0; r < 8; ++r) {
        const int m = row0 + wm * 32 + i * 16 + hi * 8 + r;
        out[(size_t)m * HID + n] = acc[i][j][r] + bv;
      }
    }
}

// ---------------------------------------------------------------- launcher
extern "C" void kernel_launch(void* const* d_in, const int* in_sizes, int n_in,
                              void* d_out, int out_size, void* d_ws, size_t ws_size,
                              hipStream_t stream) {
  const float* x     = (const float*)d_in[0];
  const float* Wqkv  = (const float*)d_in[1];
  const float* bqkv  = (const float*)d_in[2];
  const float* Wproj = (const float*)d_in[3];
  const float* bproj = (const float*)d_in[4];
  float* out = (float*)d_out;

  const size_t headElems = (size_t)BATCH * NH * SEQ * HD;  // 33.5M f16 each
  _Float16* qh     = (_Float16*)d_ws;
  _Float16* kh     = qh + headElems;
  _Float16* vh     = kh + headElems;
  _Float16* ch     = vh + headElems;
  _Float16* WqkvT  = ch + headElems;
  _Float16* WprojT = WqkvT + (size_t)N3 * HID;
  _Float16* xh     = WprojT + (size_t)HID * HID;

  convert_x<<<dim3((BATCH * SEQ * HID) / (256 * 8)), 256, 0, stream>>>(x, xh);
  convert_transpose<<<dim3(N3 / 32, HID / 32), dim3(32, 8), 0, stream>>>(
      Wqkv, WqkvT, HID, N3);
  convert_transpose<<<dim3(HID / 32, HID / 32), dim3(32, 8), 0, stream>>>(
      Wproj, WprojT, HID, HID);
  qkv_gemm<<<dim3(N3 / 128, (BATCH * SEQ) / 128), 256, 0, stream>>>(
      xh, WqkvT, bqkv, qh, kh, vh);
  attn<<<dim3(SEQ / 128, NH, BATCH), 256, 0, stream>>>(qh, kh, vh, ch);
  proj_gemm<<<dim3(HID / 128, (BATCH * SEQ) / 128), 256, 0, stream>>>(
      ch, WprojT, bproj, out);
}
